// Model_39402029973680
// MI455X (gfx1250) — compile-verified, matrix operand order checked by reading, not run on hardware
//
#include <hip/hip_runtime.h>
#include <hip/hip_bf16.h>

// ---------------------------------------------------------------------------
// GRU decoder, persistent-kernel formulation for MI455X (gfx1250).
//   - 20,100 strictly sequential GRU cells  -> latency-bound.
//   - W_hh (bf16) split across NWG=8 WGPs, resident in LDS (192 KB/WGP).
//   - hp = h @ W_hh.T via v_wmma_f32_16x16x32_bf16 (12 waves x 16 k-steps).
//   - per-cell cross-WGP h exchange: packed bf16 A-tiles in L2 (double
//     buffered) + agent-scope atomic barrier with s_sleep spin.
//   - h staging uses gfx1250 async global->LDS loads (ASYNCcnt) when the
//     toolchain exposes the builtin; falls back to a VGPR bounce copy.
// ---------------------------------------------------------------------------

#define H      512
#define TH3    1536
#define BATCH  8
#define TLEN   201
#define NCHAR  113
#define NWG    8
#define BLK    384            // 12 waves (wave32)
#define HC     64             // hidden columns owned per WGP
#define RWS    192            // hp rows per WGP  (3 gates * HC)
#define NTILE  12             // RWS / 16
#define KTILES 16             // H / 32

// ---- dynamic LDS layout (bytes) ----
#define SW_OFF    0                          // packed W_hh B-tiles: 12*16*1024 = 196608
#define SH_OFF    196608                     // staged h A-tiles (16 KB), reused as f32 h in xproj phase
#define SHP_OFF   (196608 + 16384)           // hp tiles: 12*256 f32 = 12288
#define SH32_OFF  (SHP_OFF + 12288)          // f32 hidden slice [8][64] = 2048
#define SBHH_OFF  (SH32_OFF + 2048)          // b_hh slice 192 f32
#define SBIH_OFF  (SBHH_OFF + 768)           // b_ih slice 192 f32
#define SMEM_BYTES (SBIH_OFF + 768)          // 228864 B  (< 320 KB WGP LDS)

// ---- workspace layout (bytes) ----
#define WS_CNT    0
#define WS_HBUF   128                         // 2 x 16384 B packed bf16 h (ping-pong)
#define WS_HPLAIN (WS_HBUF + 32768)           // f32 h [8][512]
#define WS_SEQ    (WS_HPLAIN + 16384)         // f32 [T][8][512]
#define WS_XPROJ  (WS_SEQ + (TLEN*BATCH*H*4)) // f32 [T][8][1536]

typedef __bf16 v16bf __attribute__((ext_vector_type(16)));
typedef float  v8f   __attribute__((ext_vector_type(8)));
typedef int    v4i   __attribute__((vector_size(16)));   // matches builtin param type

#if defined(__gfx1250__) && __has_builtin(__builtin_amdgcn_global_load_async_to_lds_b128)
#define USE_ASYNC_LDS 1
typedef __attribute__((address_space(1))) v4i* gptr_v4i;  // global int4*
typedef __attribute__((address_space(3))) v4i* lptr_v4i;  // LDS int4*
#else
#define USE_ASYNC_LDS 0
#endif

__device__ __forceinline__ unsigned short f2bf(float f) {
  unsigned u = __float_as_uint(f);
  unsigned r = u + 0x7fffu + ((u >> 16) & 1u);   // round-to-nearest-even
  return (unsigned short)(r >> 16);
}

// ushort index of element (row b, col gcol) inside the packed 16xH A layout
// (16 tiles of 16x32 bf16, per CDNA5 16-bit A-matrix VGPR layout).
__device__ __forceinline__ int abf_idx(int b, int gcol) {
  int kt = gcol >> 5, kk = gcol & 31;
  int ln = b + (((kk & 15) >= 8) ? 16 : 0);
  int vg = ((kk >= 16) ? 4 : 0) + ((kk & 7) >> 1);
  int hf = kk & 1;
  return kt * 512 + ln * 16 + vg * 2 + hf;
}

__device__ __forceinline__ float sigm_(float x) {
  return 1.0f / (1.0f + __expf(-x));
}
__device__ __forceinline__ float tanh_(float x) {
  float a = fabsf(x);
  float e = __expf(-2.0f * a);
  float r = (1.0f - e) / (1.0f + e);
  return copysignf(r, x);
}

__device__ __forceinline__ void gbar(unsigned* counter, unsigned& barnum) {
  __threadfence();
  __syncthreads();
  if (threadIdx.x == 0) {
    __hip_atomic_fetch_add(counter, 1u, __ATOMIC_RELEASE, __HIP_MEMORY_SCOPE_AGENT);
    const unsigned tgt = (barnum + 1u) * (unsigned)NWG;
    while (__hip_atomic_load(counter, __ATOMIC_ACQUIRE, __HIP_MEMORY_SCOPE_AGENT) < tgt)
      __builtin_amdgcn_s_sleep(2);
  }
  __syncthreads();
  ++barnum;
}

// Stage 16 KB of packed h (A-tiles) from L2 into LDS.
__device__ __forceinline__ void stage_h(const unsigned short* gsrc,
                                        unsigned short* ldst, int tid) {
#if USE_ASYNC_LDS
  for (int i = tid; i < 1024; i += BLK) {
    gptr_v4i g = (gptr_v4i)(unsigned long long)((const v4i*)gsrc + i);
    lptr_v4i l = (lptr_v4i)(unsigned)(unsigned long long)((v4i*)ldst + i);
    __builtin_amdgcn_global_load_async_to_lds_b128(g, l, 0, 0);
  }
#if __has_builtin(__builtin_amdgcn_s_wait_asynccnt)
  __builtin_amdgcn_s_wait_asynccnt(0);
#else
  asm volatile("s_wait_asynccnt 0x0" ::: "memory");
#endif
#else
  const uint4* src = (const uint4*)gsrc;
  uint4* dst = (uint4*)ldst;
  for (int i = tid; i < 1024; i += BLK) dst[i] = src[i];
#endif
}

__global__ void __launch_bounds__(BLK, 1)
gru_persistent(const float* __restrict__ feat, const float* __restrict__ embed,
               const float* __restrict__ W_ih, const float* __restrict__ W_hh,
               const float* __restrict__ b_ih, const float* __restrict__ b_hh,
               unsigned* counter, unsigned short* hbuf,
               float* hplain, float* seq, float* xproj)
{
  extern __shared__ char smem[];
  unsigned short* sW  = (unsigned short*)(smem + SW_OFF);
  unsigned short* sH  = (unsigned short*)(smem + SH_OFF);
  float* sHf  = (float*)(smem + SH_OFF);     // reuse as f32 h in xproj phase
  float* sHP  = (float*)(smem + SHP_OFF);
  float* sh32 = (float*)(smem + SH32_OFF);
  float* sbhh = (float*)(smem + SBHH_OFF);
  float* sbih = (float*)(smem + SBIH_OFF);

  const int tid  = threadIdx.x;
  const int wg   = blockIdx.x;               // 0..7 : owns hidden cols [wg*64, wg*64+64)
  const int wave = tid >> 5;
  const int lane = tid & 31;
  unsigned barnum = 0;
  unsigned cellidx = 0;                      // global cell index (for ping-pong)

  // ------------------------- one-time setup -------------------------
  // Pack my 192 W_hh rows into CDNA5 B-tile layout in LDS (bf16).
  for (int idx = tid; idx < RWS * H; idx += BLK) {
    int R = idx >> 9, col = idx & (H - 1);
    int g = R >> 6, c = R & (HC - 1);
    int gr = g * H + wg * HC + c;            // global W_hh row (r/z/n interleave)
    int kt = col >> 5, kk = col & 31;
    int N = R & 15, n = R >> 4;
    int ln = N + (((kk & 15) >= 8) ? 16 : 0);
    int vg = ((kk >= 16) ? 4 : 0) + ((kk & 7) >> 1);
    int hf = kk & 1;
    sW[(n * KTILES + kt) * 512 + ln * 16 + vg * 2 + hf] =
        f2bf(W_hh[(long)gr * H + col]);
  }
  for (int i = tid; i < RWS; i += BLK) {
    int g = i >> 6, c = i & (HC - 1);
    int gr = g * H + wg * HC + c;
    sbhh[i] = b_hh[gr];
    sbih[i] = b_ih[gr];
  }
  // Initial packed h (buffer 0) = feat; zero-pad rows 8..15 (both buffers).
  for (int e = tid; e < 16 * HC; e += BLK) {
    int b = e >> 6, c = e & (HC - 1);
    int gcol = wg * HC + c;
    float v = (b < BATCH) ? feat[b * H + gcol] : 0.0f;
    hbuf[abf_idx(b, gcol)] = f2bf(v);
    if (b >= BATCH) hbuf[8192 + abf_idx(b, gcol)] = f2bf(0.0f);
  }
  // f32 hidden slice + seq[0] = <SOS> embedding broadcast.
  for (int e = tid; e < BATCH * HC; e += BLK) {
    int b = e >> 6, c = e & (HC - 1);
    sh32[e] = feat[b * H + wg * HC + c];
    seq[(long)b * H + wg * HC + c] = embed[wg * HC + c];
  }
  // xproj[0] = sos @ W_ih.T + b_ih  (same for every batch row).
  for (int i = tid; i < RWS; i += BLK) {
    int g = i >> 6, c = i & (HC - 1);
    int gr = g * H + wg * HC + c;
    const float* wrow = W_ih + (long)gr * H;
    float acc = b_ih[gr];
    for (int kk = 0; kk < H; ++kk) acc += wrow[kk] * embed[kk];
    for (int b = 0; b < BATCH; ++b) xproj[(long)b * TH3 + gr] = acc;
  }
  gbar(counter, barnum);                     // all WGPs ready

  // ------------------------- main recurrence -------------------------
  for (int t = 0; t < TLEN - 1; ++t) {
    for (int k = 0; k <= t; ++k) {
      const int roff = (cellidx & 1u) * 8192;        // ushort offsets
      const int woff = 8192 - roff;

      // Stage packed h (16 KB) from L2 into LDS A-tiles (async on gfx1250).
      stage_h(hbuf + roff, sH, tid);

      // Early (latency-hidden) loads of this cell's input projections.
      float xr[2], xz[2], xn[2];
#pragma unroll
      for (int s = 0; s < 2; ++s) {
        int e = tid + s * BLK;
        if (e < BATCH * HC) {
          int b = e >> 6, c = e & (HC - 1);
          long base = ((long)k * BATCH + b) * TH3;
          int rr = wg * HC + c;
          xr[s] = xproj[base + rr];
          xz[s] = xproj[base + H + rr];
          xn[s] = xproj[base + 2 * H + rr];
          if (k < t) {                               // prefetch next cell's rows
            __builtin_prefetch(&xproj[base + (long)BATCH * TH3 + rr], 0, 1);
          }
        } else { xr[s] = xz[s] = xn[s] = 0.0f; }
      }
      __syncthreads();

      // ---- WMMA phase: wave computes its 16-row hp tile over K=512 ----
      v8f acc0 = {}; v8f acc1 = {};
      const char* aw = (const char*)sH + lane * 32;
      const char* bw = (const char*)sW + (long)wave * KTILES * 1024 + lane * 32;
#pragma unroll
      for (int kt = 0; kt < KTILES; kt += 2) {
        v16bf a0 = *(const v16bf*)(aw + kt * 1024);
        v16bf b0 = *(const v16bf*)(bw + kt * 1024);
        acc0 = __builtin_amdgcn_wmma_f32_16x16x32_bf16(false, a0, false, b0,
                                                       (short)0, acc0, false, false);
        v16bf a1 = *(const v16bf*)(aw + (kt + 1) * 1024);
        v16bf b1 = *(const v16bf*)(bw + (kt + 1) * 1024);
        acc1 = __builtin_amdgcn_wmma_f32_16x16x32_bf16(false, a1, false, b1,
                                                       (short)0, acc1, false, false);
      }
      v8f accs = acc0 + acc1;
      *(v8f*)(sHP + wave * 256 + lane * 8) = accs;   // [tile][lane][vgpr]
      __syncthreads();

      // ---- gate phase: 512 (b,c) elements over 384 threads ----
#pragma unroll
      for (int s = 0; s < 2; ++s) {
        int e = tid + s * BLK;
        if (e >= BATCH * HC) break;
        int b = e >> 6, c = e & (HC - 1);
        int Rr = c, Rz = HC + c, Rn = 2 * HC + c;
        float hr = sHP[(Rr >> 4) * 256 + (Rr & 15) * 8 + b] + sbhh[Rr];
        float hz = sHP[(Rz >> 4) * 256 + (Rz & 15) * 8 + b] + sbhh[Rz];
        float hn = sHP[(Rn >> 4) * 256 + (Rn & 15) * 8 + b] + sbhh[Rn];
        float r  = sigm_(xr[s] + hr);
        float z  = sigm_(xz[s] + hz);
        float nn = tanh_(xn[s] + r * hn);
        float hprev = sh32[e];
        float hnew  = (1.0f - z) * nn + z * hprev;
        sh32[e] = hnew;
        int gcol = wg * HC + c;
        hbuf[woff + abf_idx(b, gcol)] = f2bf(hnew);
        if (k == t) {                                // h_fin of this outer step
          hplain[(long)b * H + gcol] = hnew;
          seq[((long)(t + 1) * BATCH + b) * H + gcol] = hnew;
        }
      }
      ++cellidx;
      gbar(counter, barnum);                         // device-wide cell barrier
    }

    // ---- input-side projection for position t+1 (WGP-private rows) ----
    for (int i = tid; i < BATCH * H; i += BLK) sHf[i] = hplain[i];
    __syncthreads();
    for (int i = tid; i < RWS * BATCH; i += BLK) {
      int R = i % RWS, b = i / RWS;
      int g = R >> 6, c = R & (HC - 1);
      int gr = g * H + wg * HC + c;
      const float* wrow = W_ih + (long)gr * H;
      const float* hv = sHf + b * H;
      float acc = sbih[R];
      for (int kk = 0; kk < H; ++kk) acc += wrow[kk] * hv[kk];
      xproj[((long)(t + 1) * BATCH + b) * TH3 + gr] = acc;
    }
    __syncthreads();                                 // sHf reused next cell
  }
}

// out[b][ch][t] = seq[t][b][:] . proj_W[ch][:] + proj_b[ch]
__global__ void __launch_bounds__(128)
proj_kernel(const float* __restrict__ seq, const float* __restrict__ projW,
            const float* __restrict__ projb, float* __restrict__ out)
{
  __shared__ float sv[H];
  int tb = blockIdx.x;                 // 0 .. T*B-1
  int t = tb >> 3, b = tb & 7;
  const float* src = seq + ((long)t * BATCH + b) * H;
  for (int i = threadIdx.x; i < H; i += blockDim.x) sv[i] = src[i];
  __syncthreads();
  for (int ch = threadIdx.x; ch < NCHAR; ch += blockDim.x) {
    const float* w = projW + (long)ch * H;
    float acc = projb[ch];
    for (int i = 0; i < H; ++i) acc += w[i] * sv[i];
    out[((long)b * NCHAR + ch) * TLEN + t] = acc;
  }
}

extern "C" void kernel_launch(void* const* d_in, const int* in_sizes, int n_in,
                              void* d_out, int out_size, void* d_ws, size_t ws_size,
                              hipStream_t stream) {
  (void)in_sizes; (void)n_in; (void)out_size; (void)ws_size;
  const float* feat   = (const float*)d_in[0];
  const float* embed  = (const float*)d_in[1];
  const float* W_ih   = (const float*)d_in[2];
  const float* W_hh   = (const float*)d_in[3];
  const float* b_ih   = (const float*)d_in[4];
  const float* b_hh   = (const float*)d_in[5];
  const float* proj_W = (const float*)d_in[6];
  const float* proj_b = (const float*)d_in[7];

  char* ws = (char*)d_ws;
  unsigned*       counter = (unsigned*)(ws + WS_CNT);
  unsigned short* hbuf    = (unsigned short*)(ws + WS_HBUF);
  float*          hplain  = (float*)(ws + WS_HPLAIN);
  float*          seq     = (float*)(ws + WS_SEQ);
  float*          xproj   = (float*)(ws + WS_XPROJ);

  (void)hipMemsetAsync(counter, 0, 128, stream);     // deterministic barrier state
  (void)hipFuncSetAttribute((const void*)gru_persistent,
                            hipFuncAttributeMaxDynamicSharedMemorySize, SMEM_BYTES);

  hipLaunchKernelGGL(gru_persistent, dim3(NWG), dim3(BLK), SMEM_BYTES, stream,
                     feat, embed, W_ih, W_hh, b_ih, b_hh,
                     counter, hbuf, hplain, seq, xproj);

  hipLaunchKernelGGL(proj_kernel, dim3(TLEN * BATCH), dim3(128), 0, stream,
                     seq, proj_W, proj_b, (float*)d_out);
}
